// Attention_22265110462645
// MI455X (gfx1250) — matrix operand order repaired and checked by reading, not compiled, not run on hardware
//
#include <hip/hip_runtime.h>

// ---------------------------------------------------------------------------
// GPT-2 attention block for MI455X (gfx1250, wave32, WMMA f16 -> f32 accum)
// B=8, S=1024, NX=1024, H=16, D=64
// ---------------------------------------------------------------------------

typedef _Float16 h16 __attribute__((ext_vector_type(16)));
typedef _Float16 h8  __attribute__((ext_vector_type(8)));
typedef _Float16 h4  __attribute__((ext_vector_type(4)));
typedef _Float16 h2  __attribute__((ext_vector_type(2)));
typedef float    f8  __attribute__((ext_vector_type(8)));
typedef float    f4  __attribute__((ext_vector_type(4)));

static __device__ __forceinline__ h8 ldh8(const _Float16* p) {
  return *reinterpret_cast<const h8*>(p);
}
static __device__ __forceinline__ h16 cat(h8 lo, h8 hi) {
  return __builtin_shufflevector(lo, hi, 0,1,2,3,4,5,6,7,8,9,10,11,12,13,14,15);
}
static __device__ __forceinline__ f8 wmma16(h16 a, h16 b, f8 c) {
  // v_wmma_f32_16x16x32_f16
  return __builtin_amdgcn_wmma_f32_16x16x32_f16(false, a, false, b, (short)0, c,
                                                false, false);
}
// v_cvt_pk_rtz_f16_f32 with the builtin's __fp16 vector bit-cast to _Float16
static __device__ __forceinline__ h2 pkrtz(float a, float b) {
  return __builtin_bit_cast(h2, __builtin_amdgcn_cvt_pkrtz(a, b));
}

// 16-lane row reduction with DPP (no LDS): xor1, xor2 via quad_perm, then
// row_half_mirror (combines quads within 8), row_mirror (combines 8-halves).
template <int CTRL>
static __device__ __forceinline__ float dppf(float x) {
  return __builtin_bit_cast(
      float, __builtin_amdgcn_update_dpp(0, __builtin_bit_cast(int, x), CTRL,
                                         0xf, 0xf, true));
}
static __device__ __forceinline__ float rmax16(float v) {
  v = fmaxf(v, dppf<0xB1>(v));   // quad_perm [1,0,3,2]  (xor 1)
  v = fmaxf(v, dppf<0x4E>(v));   // quad_perm [2,3,0,1]  (xor 2)
  v = fmaxf(v, dppf<0x141>(v));  // row_half_mirror      (xor 7 within 8)
  v = fmaxf(v, dppf<0x140>(v));  // row_mirror           (xor 15 within 16)
  return v;
}
static __device__ __forceinline__ float rsum16(float v) {
  v += dppf<0xB1>(v);
  v += dppf<0x4E>(v);
  v += dppf<0x141>(v);
  v += dppf<0x140>(v);
  return v;
}

// load 8 elements and convert to f16 (pk_rtz) if needed
template <typename TA>
static __device__ __forceinline__ h8 load_cvt8(const TA* p) {
  if constexpr (sizeof(TA) == 2) {
    return *reinterpret_cast<const h8*>(p);
  } else {
    f4 x = *reinterpret_cast<const f4*>(p);
    f4 y = *reinterpret_cast<const f4*>(p + 4);
    h2 p0 = pkrtz(x[0], x[1]);
    h2 p1 = pkrtz(x[2], x[3]);
    h2 p2 = pkrtz(y[0], y[1]);
    h2 p3 = pkrtz(y[2], y[3]);
    h4 lo = __builtin_shufflevector(p0, p1, 0, 1, 2, 3);
    h4 hi = __builtin_shufflevector(p2, p3, 0, 1, 2, 3);
    return __builtin_shufflevector(lo, hi, 0, 1, 2, 3, 4, 5, 6, 7);
  }
}

// ---------------------------------------------------------------------------
// Generic GEMM + bias:  C[M,N] = A[M,K] (f32 or f16) * W[K,N] (f32) + bias
// 128x128 block tile, k-step 32, 8 waves each computing 64x32 (4x2 WMMA tiles)
// ---------------------------------------------------------------------------
template <typename TA, typename TO>
__global__ __launch_bounds__(256) void gemm_bias(
    const TA* __restrict__ A, const float* __restrict__ W,
    const float* __restrict__ bias, TO* __restrict__ C,
    int M, int N, int K) {
  __shared__ __align__(16) _Float16 As[128][32];   // [m][k]
  __shared__ __align__(16) _Float16 Bs[128][32];   // [n][k]  (W transposed)

  const int tid  = threadIdx.x;
  const int lane = tid & 31;
  const int wid  = tid >> 5;
  const int ln   = lane & 15;
  const int hs   = lane >> 4;
  const int gm = blockIdx.x * 128;
  const int gn = blockIdx.y * 128;
  const int wm = (wid >> 2) * 64;   // 0 or 64
  const int wn = (wid & 3) * 32;    // 0,32,64,96

  f8 acc[4][2];
  #pragma unroll
  for (int i = 0; i < 4; ++i)
    #pragma unroll
    for (int j = 0; j < 2; ++j) acc[i][j] = f8{};

  float bn[2];
  #pragma unroll
  for (int j = 0; j < 2; ++j) bn[j] = bias[gn + wn + j * 16 + ln];

  const int ar = tid >> 2;          // 0..63  (A rows; also +64)
  const int ac = (tid & 3) * 8;     // 0,8,16,24
  const int wk = (tid & 15) * 2;    // 0..30  (W k-pair)
  const int wc = (tid >> 4) * 8;    // 0..120 (W col group of 8)

  for (int kk = 0; kk < K; kk += 32) {
    __syncthreads();  // previous iteration's readers done
    // A tile (convert to f16), b128 stores
    #pragma unroll
    for (int rr = 0; rr < 2; ++rr) {
      int r = ar + rr * 64;
      const TA* ap = A + (size_t)(gm + r) * K + kk + ac;
      *reinterpret_cast<h8*>(&As[r][ac]) = load_cvt8(ap);
    }
    // W tile, stored transposed [n][k]: pack k-pairs -> ds_store_b32
    {
      const float* w0 = W + (size_t)(kk + wk) * N + gn + wc;
      const float* w1 = w0 + N;
      f4 a = *reinterpret_cast<const f4*>(w0);
      f4 b2 = *reinterpret_cast<const f4*>(w0 + 4);
      f4 c = *reinterpret_cast<const f4*>(w1);
      f4 d2 = *reinterpret_cast<const f4*>(w1 + 4);
      #pragma unroll
      for (int i = 0; i < 4; ++i) {
        *reinterpret_cast<h2*>(&Bs[wc + i][wk])     = pkrtz(a[i], c[i]);
        *reinterpret_cast<h2*>(&Bs[wc + 4 + i][wk]) = pkrtz(b2[i], d2[i]);
      }
    }
    __syncthreads();

    h16 af[4];
    #pragma unroll
    for (int i = 0; i < 4; ++i) {
      const _Float16* p = &As[wm + i * 16 + ln][0];
      // A layout: lo8 -> K = hs*8.., hi8 -> K = 16+hs*8..
      af[i] = cat(ldh8(p + hs * 8), ldh8(p + 16 + hs * 8));
    }
    h16 bf[2];
    #pragma unroll
    for (int j = 0; j < 2; ++j) {
      const _Float16* p = &Bs[wn + j * 16 + ln][0];
      // B layout: lo8 -> K = hs*8.., hi8 -> K = 16+hs*8..
      bf[j] = cat(ldh8(p + hs * 8), ldh8(p + 16 + hs * 8));
    }
    #pragma unroll
    for (int i = 0; i < 4; ++i)
      #pragma unroll
      for (int j = 0; j < 2; ++j)
        acc[i][j] = wmma16(af[i], bf[j], acc[i][j]);
  }

  // epilogue: C/D layout: reg r -> row r+8*hs, lane -> col ln
  #pragma unroll
  for (int i = 0; i < 4; ++i)
    #pragma unroll
    for (int j = 0; j < 2; ++j)
      #pragma unroll
      for (int r = 0; r < 8; ++r) {
        int m = gm + wm + i * 16 + r + 8 * hs;
        int n = gn + wn + j * 16 + ln;
        C[(size_t)m * N + n] = (TO)(acc[i][j][r] + bn[j]);
      }
}

// ---------------------------------------------------------------------------
// Causal flash attention: block = (b,h, 128 q rows), 8 waves x 16 q rows.
// qkv: f16 [B*S][3*NX]; aout: f16 [B*S][NX] (heads merged)
// K tile staged via CDNA5 async global->LDS DMA; V staged transposed.
// ---------------------------------------------------------------------------
__global__ __launch_bounds__(256) void attn_fa(
    const _Float16* __restrict__ qkv, _Float16* __restrict__ aout) {
  constexpr int S = 1024, NX = 1024, TR = 3072;
  __shared__ __align__(16) _Float16 Ks[32][64];      // [key][d]
  __shared__ __align__(16) _Float16 Vs[64][32];      // [d][key] (transposed)
  __shared__ __align__(16) _Float16 Ps[8][16][32];   // per-wave P relayout

  const int tid  = threadIdx.x;
  const int lane = tid & 31;
  const int wid  = tid >> 5;
  const int ln   = lane & 15;
  const int hs   = lane >> 4;
  const int bh = blockIdx.x;
  const int b  = bh >> 4;
  const int h  = bh & 15;
  const int qb = blockIdx.y * 128;
  const int qrow0 = qb + wid * 16;

  // resident Q fragments (A layout), pre-scaled by 1/sqrt(D) = 1/8
  const _Float16* qp = qkv + (size_t)(b * S + qrow0 + ln) * TR + h * 64;
  h16 qf0 = cat(ldh8(qp + hs * 8),      ldh8(qp + 16 + hs * 8));
  h16 qf1 = cat(ldh8(qp + 32 + hs * 8), ldh8(qp + 48 + hs * 8));
  qf0 = qf0 * (_Float16)0.125f;
  qf1 = qf1 * (_Float16)0.125f;

  f8 o[4];
  #pragma unroll
  for (int j = 0; j < 4; ++j) o[j] = f8{};
  float mrow[8], lrow[8];
  #pragma unroll
  for (int r = 0; r < 8; ++r) { mrow[r] = -1e30f; lrow[r] = 0.0f; }

  const int nchunk = qb / 32 + 4;        // causal bound for this block

  // K staging: one async b128 per thread (32 keys x 64 d)
  const int kvr = tid >> 3;              // 0..31 key row
  const int kvd = (tid & 7) * 8;         // 0..56 d offset
  const unsigned ks_lds = (unsigned)(uintptr_t)&Ks[kvr][kvd];
  // V staging: key-pairs packed along LDS rows of Vs[d][key]
  const int vk = (tid & 15) * 2;         // 0..30 key pair
  const int vd = (tid >> 4) * 4;         // 0..60 d group

  for (int ck = 0; ck < nchunk; ++ck) {
    const int key0 = ck * 32;
    __syncthreads();  // all readers of Ks/Vs done
    {
      // K: async DMA global -> LDS (CDNA5), tracked by ASYNCcnt
      const _Float16* kp =
          qkv + (size_t)(b * S + key0 + kvr) * TR + NX + h * 64 + kvd;
      asm volatile("global_load_async_to_lds_b128 %0, %1, off"
                   :: "v"(ks_lds),
                      "v"((unsigned long long)(uintptr_t)kp)
                   : "memory");
      // V: load 2 keys x 4 d, store transposed as packed pairs
      const _Float16* vp =
          qkv + (size_t)(b * S + key0 + vk) * TR + 2 * NX + h * 64 + vd;
      h4 a0 = *reinterpret_cast<const h4*>(vp);
      h4 a1 = *reinterpret_cast<const h4*>(vp + TR);
      #pragma unroll
      for (int i = 0; i < 4; ++i) {
        h2 pr = {a0[i], a1[i]};
        *reinterpret_cast<h2*>(&Vs[vd + i][vk]) = pr;
      }
      asm volatile("s_wait_asynccnt 0x0" ::: "memory");
    }
    __syncthreads();

    // waves skip chunks that are entirely above the causal diagonal
    if (key0 <= qrow0 + 15) {
      // scores: S = Q @ K^T, two 16x16 tiles (keys 0..15, 16..31)
      f8 s[2];
      #pragma unroll
      for (int t = 0; t < 2; ++t) {
        const _Float16* kr = &Ks[t * 16 + ln][0];
        h16 kf0 = cat(ldh8(kr + hs * 8),      ldh8(kr + 16 + hs * 8));
        h16 kf1 = cat(ldh8(kr + 32 + hs * 8), ldh8(kr + 48 + hs * 8));
        f8 a = f8{};
        a = wmma16(qf0, kf0, a);
        a = wmma16(qf1, kf1, a);
        s[t] = a;
      }

      float v0[8], v1[8];
      #pragma unroll
      for (int r = 0; r < 8; ++r) { v0[r] = s[0][r]; v1[r] = s[1][r]; }
      if (key0 + 31 > qrow0) {  // wave-uniform: only diagonal chunks mask
        #pragma unroll
        for (int r = 0; r < 8; ++r) {
          const int q = qrow0 + r + 8 * hs;
          if (key0 + ln > q)      v0[r] = -1e30f;
          if (key0 + 16 + ln > q) v1[r] = -1e30f;
        }
      }

      // online softmax over the 32-key chunk (DPP row reductions)
      float p0[8], p1[8], corr[8];
      #pragma unroll
      for (int r = 0; r < 8; ++r) {
        float tmax = rmax16(fmaxf(v0[r], v1[r]));
        float nm = fmaxf(mrow[r], tmax);
        float c  = __expf(mrow[r] - nm);
        mrow[r] = nm;
        p0[r] = __expf(v0[r] - nm);
        p1[r] = __expf(v1[r] - nm);
        float rs = rsum16(p0[r] + p1[r]);
        lrow[r] = lrow[r] * c + rs;
        corr[r] = c;
      }
      #pragma unroll
      for (int j = 0; j < 4; ++j)
        #pragma unroll
        for (int r = 0; r < 8; ++r) o[j][r] *= corr[r];

      // relayout P (C layout -> A layout) via per-wave LDS tile.
      // Same-wave DS ops are in-order (ISA ch.11); no block barrier needed.
      #pragma unroll
      for (int r = 0; r < 8; ++r) {
        const int lr = r + 8 * hs;
        Ps[wid][lr][ln]      = (_Float16)p0[r];
        Ps[wid][lr][16 + ln] = (_Float16)p1[r];
      }
      __builtin_amdgcn_wave_barrier();
      const _Float16* pr = &Ps[wid][ln][0];
      const h16 pA = cat(ldh8(pr + hs * 8), ldh8(pr + 16 + hs * 8));

      // O += P @ V   (B frag: lane = d col, keys hs*8.., 16+hs*8.. from Vs[d][*])
      #pragma unroll
      for (int j = 0; j < 4; ++j) {
        const _Float16* vr = &Vs[j * 16 + ln][0];
        h16 vB = cat(ldh8(vr + hs * 8), ldh8(vr + 16 + hs * 8));
        o[j] = wmma16(pA, vB, o[j]);
      }
    }
  }

  // normalize & store merged-head output
  #pragma unroll
  for (int r = 0; r < 8; ++r) {
    const float inv = __builtin_amdgcn_rcpf(lrow[r]);
    const int q = qrow0 + r + 8 * hs;
    #pragma unroll
    for (int j = 0; j < 4; ++j) {
      const int d = j * 16 + ln;
      aout[(size_t)(b * S + q) * NX + h * 64 + d] = (_Float16)(o[j][r] * inv);
    }
  }
}

// ---------------------------------------------------------------------------
extern "C" void kernel_launch(void* const* d_in, const int* in_sizes, int n_in,
                              void* d_out, int out_size, void* d_ws,
                              size_t ws_size, hipStream_t stream) {
  (void)in_sizes; (void)n_in; (void)out_size; (void)ws_size;
  const float* x      = (const float*)d_in[0];
  const float* w_attn = (const float*)d_in[1];
  const float* b_attn = (const float*)d_in[2];
  const float* w_proj = (const float*)d_in[3];
  const float* b_proj = (const float*)d_in[4];
  float* out = (float*)d_out;

  _Float16* qkv  = (_Float16*)d_ws;                  // [8192][3072] f16 (48 MB)
  _Float16* abuf = qkv + (size_t)8192 * 3072;        // [8192][1024] f16 (16 MB)

  dim3 blk(256);
  // 1) fused QKV projection
  gemm_bias<float, _Float16><<<dim3(64, 24), blk, 0, stream>>>(
      x, w_attn, b_attn, qkv, 8192, 3072, 1024);
  // 2) causal flash attention per (b,h)
  attn_fa<<<dim3(128, 8), blk, 0, stream>>>(qkv, abuf);
  // 3) output projection
  gemm_bias<_Float16, float><<<dim3(64, 8), blk, 0, stream>>>(
      abuf, w_proj, b_proj, out, 8192, 1024, 1024);
}
